// FlexAttentionAPI_29764123361366
// MI455X (gfx1250) — matrix operand order repaired and checked by reading, not compiled, not run on hardware
//
#include <hip/hip_runtime.h>
#include <hip/hip_bf16.h>
#include <math.h>

#define EMBED 2048
#define HEADS 16
#define HDIM  128
#define BATCH 2
#define SEQ   2048
#define MROWS (BATCH * SEQ)   // 4096

typedef __bf16 bf16;
typedef __bf16 v16bf __attribute__((ext_vector_type(16)));
typedef __bf16 v8bf  __attribute__((ext_vector_type(8)));
typedef __bf16 v4bf  __attribute__((ext_vector_type(4)));
typedef float  v8f   __attribute__((ext_vector_type(8)));

// ---------------------------------------------------------------------------
// WMMA helpers (CDNA5 wave32, V_WMMA_F32_16X16X32_BF16)
// ---------------------------------------------------------------------------
__device__ __forceinline__ v8f wmma_bf16(v16bf a, v16bf b, v8f c) {
  return __builtin_amdgcn_wmma_f32_16x16x32_bf16(
      /*neg_a=*/false, a, /*neg_b=*/false, b,
      /*c_mod=*/(short)0, c, /*reuse_a=*/false, /*reuse_b=*/false);
}

// A fragment: 16x32 (MxK), source row-major [row][k], element stride `ld`.
// Lane (0-15): row=lane, K in [hi*8, hi*8+8) -> VGPR0-3, [16+hi*8, +8) -> VGPR4-7.
__device__ __forceinline__ v16bf ld_frag_a(const bf16* base, int ld, int lane) {
  const int row = lane & 15;
  const int hi  = lane >> 4;
  const bf16* p = base + row * ld + hi * 8;
  v8bf lo = *(const v8bf*)(p);
  v8bf up = *(const v8bf*)(p + 16);
  v16bf r;
#pragma unroll
  for (int i = 0; i < 8; ++i) { r[i] = lo[i]; r[i + 8] = up[i]; }
  return r;
}

// B fragment: 32x16 (KxN), source stored row-major as [n][k] (W-style),
// element stride `ld`. Lane: n=lane&15, K in [hi*8, hi*8+8) -> VGPR0-3, [16+hi*8, +8) -> VGPR4-7.
__device__ __forceinline__ v16bf ld_frag_b(const bf16* base, int ld, int lane) {
  const int n  = lane & 15;
  const int hi = lane >> 4;
  const bf16* p = base + n * ld + hi * 8;
  v8bf lo = *(const v8bf*)(p);
  v8bf up = *(const v8bf*)(p + 16);
  v16bf r;
#pragma unroll
  for (int i = 0; i < 8; ++i) { r[i] = lo[i]; r[i + 8] = up[i]; }
  return r;
}

// ---------------------------------------------------------------------------
// f32 -> bf16 conversion (4-wide)
// ---------------------------------------------------------------------------
__global__ void cvt_f32_bf16(const float* __restrict__ in,
                             bf16* __restrict__ out, int n4) {
  int i = blockIdx.x * blockDim.x + threadIdx.x;
  if (i < n4) {
    float4 f = ((const float4*)in)[i];
    v4bf r;
    r[0] = (bf16)f.x; r[1] = (bf16)f.y; r[2] = (bf16)f.z; r[3] = (bf16)f.w;
    ((v4bf*)out)[i] = r;
  }
}

// ---------------------------------------------------------------------------
// Tiled WMMA GEMM: C[M,N] = A[M,K] * W[N,K]^T + bias
// Block: 128x128 tile, 256 threads (8 waves), each wave 32x64, K-step 32.
// Double-buffered LDS (ping-pong): global loads for chunk k+1 issue before
// the WMMAs of chunk k; ds_store waits land after the math; ONE barrier/step.
// mode 0: f32 out[m*N+n] = (acc+bias)*scale
// mode 1: bf16 [B,H,S,D] scatter (Q pre-scaled / K)
// mode 2: bf16 [B,H,D,S] scatter (V transposed)
// ---------------------------------------------------------------------------
__global__ void __launch_bounds__(256)
gemm_wmma_bf16(const bf16* __restrict__ A, const bf16* __restrict__ W,
               const float* __restrict__ bias, float* __restrict__ outF,
               bf16* __restrict__ outB, int M, int N, int K,
               int mode, float scale) {
  constexpr int LDT = 40;  // 32 + 8 pad; 80B pitch, 16B aligned
  __shared__ alignas(16) bf16 As[2][128 * LDT];
  __shared__ alignas(16) bf16 Bs[2][128 * LDT];

  const int tid   = threadIdx.x;
  const int lane  = tid & 31;
  const int wave  = tid >> 5;            // 0..7
  const int waveM = (wave >> 1) * 32;    // 0,32,64,96
  const int waveN = (wave & 1) * 64;     // 0,64
  const int mBase = blockIdx.y * 128;
  const int nBase = blockIdx.x * 128;

  v8f acc[2][4];
#pragma unroll
  for (int i = 0; i < 2; ++i)
#pragma unroll
    for (int j = 0; j < 4; ++j) acc[i][j] = v8f{};

  const int ldRow = tid >> 1;          // 0..127
  const int ldCol = (tid & 1) * 16;    // 0 or 16
  const bf16* gaRow = A + (size_t)(mBase + ldRow) * K + ldCol;
  const bf16* gbRow = W + (size_t)(nBase + ldRow) * K + ldCol;
  const int ldsOff = ldRow * LDT + ldCol;

  // Prologue: fill buffer 0 with K-chunk 0
  {
    v8bf a0 = *(const v8bf*)(gaRow);
    v8bf a1 = *(const v8bf*)(gaRow + 8);
    v8bf b0 = *(const v8bf*)(gbRow);
    v8bf b1 = *(const v8bf*)(gbRow + 8);
    *(v8bf*)&As[0][ldsOff]     = a0;
    *(v8bf*)&As[0][ldsOff + 8] = a1;
    *(v8bf*)&Bs[0][ldsOff]     = b0;
    *(v8bf*)&Bs[0][ldsOff + 8] = b1;
  }
  __syncthreads();

  int cur = 0;
  for (int k0 = 0; k0 < K; k0 += 32) {
    const bool haveNext = (k0 + 32) < K;

    // Issue next chunk's global loads first (overlap with WMMAs below).
    v8bf na0, na1, nb0, nb1;
    if (haveNext) {
      const bf16* ga = gaRow + (k0 + 32);
      const bf16* gb = gbRow + (k0 + 32);
      na0 = *(const v8bf*)(ga);
      na1 = *(const v8bf*)(ga + 8);
      nb0 = *(const v8bf*)(gb);
      nb1 = *(const v8bf*)(gb + 8);
    }

    // Matrix math on the current buffer.
    v16bf af[2], bfr[4];
#pragma unroll
    for (int mt = 0; mt < 2; ++mt)
      af[mt] = ld_frag_a(&As[cur][(waveM + mt * 16) * LDT], LDT, lane);
#pragma unroll
    for (int nt = 0; nt < 4; ++nt)
      bfr[nt] = ld_frag_b(&Bs[cur][(waveN + nt * 16) * LDT], LDT, lane);
#pragma unroll
    for (int mt = 0; mt < 2; ++mt)
#pragma unroll
      for (int nt = 0; nt < 4; ++nt)
        acc[mt][nt] = wmma_bf16(af[mt], bfr[nt], acc[mt][nt]);

    // Stage next chunk into the alternate buffer (wait-for-load lands here).
    if (haveNext) {
      const int nxt = cur ^ 1;
      *(v8bf*)&As[nxt][ldsOff]     = na0;
      *(v8bf*)&As[nxt][ldsOff + 8] = na1;
      *(v8bf*)&Bs[nxt][ldsOff]     = nb0;
      *(v8bf*)&Bs[nxt][ldsOff + 8] = nb1;
    }
    __syncthreads();
    cur ^= 1;
  }

  // Epilogue: C/D layout -> lanes 0-15 hold col n, VGPR r = row r (+8 hi half)
  const int cn = lane & 15;
  const int hi = lane >> 4;
#pragma unroll
  for (int mt = 0; mt < 2; ++mt)
#pragma unroll
    for (int nt = 0; nt < 4; ++nt) {
      const int n = nBase + waveN + nt * 16 + cn;
      const float bv = bias[n];
#pragma unroll
      for (int r = 0; r < 8; ++r) {
        const int m = mBase + waveM + mt * 16 + r + hi * 8;
        const float v = (acc[mt][nt][r] + bv) * scale;
        if (mode == 0) {
          outF[(size_t)m * N + n] = v;
        } else {
          const int b = m >> 11, s = m & (SEQ - 1);      // SEQ = 2048
          const int h = n >> 7, d = n & (HDIM - 1);      // HDIM = 128
          const bf16 bvv = (bf16)v;
          if (mode == 1)
            outB[(((size_t)b * HEADS + h) * SEQ + s) * HDIM + d] = bvv;
          else
            outB[(((size_t)b * HEADS + h) * HDIM + d) * SEQ + s] = bvv;
        }
      }
    }
}

// ---------------------------------------------------------------------------
// Flash attention (causal), 4 waves / block, 64 q-rows per block,
// 64-key chunks, online softmax. Q pre-scaled by log2(e)/sqrt(D).
// Q,K: [B,H,S,D] bf16.  Vt: [B,H,D,S] bf16.  O: [B,S,E] bf16.
// Next chunk's K/V lines are prefetched (global_prefetch_b8) before compute.
// ---------------------------------------------------------------------------
__global__ void __launch_bounds__(128)
flash_wmma_bf16(const bf16* __restrict__ Q, const bf16* __restrict__ Kb,
                const bf16* __restrict__ Vt, bf16* __restrict__ O) {
  constexpr int LQK = 136;   // 128 + 8 pad (272B pitch, 16B aligned)
  constexpr int LV  = 72;    // 64 + 8 pad (144B pitch, 16B aligned)
  __shared__ alignas(16) bf16 Qs[64 * LQK];
  __shared__ alignas(16) bf16 Ks[64 * LQK];
  __shared__ alignas(16) bf16 Vs[128 * LV];
  __shared__ alignas(16) bf16 Ps[4][16 * LV];   // per-wave P staging

  const int tid  = threadIdx.x;
  const int lane = tid & 31;
  const int wave = tid >> 5;                 // 0..3
  const int qt   = blockIdx.x & (SEQ / 64 - 1);
  const int bh   = blockIdx.x >> 5;          // 0..B*H-1
  const int qTile = qt * 64;
  const size_t qkBase = (size_t)bh * SEQ * HDIM;
  const int qW = wave * 16;                  // wave's q offset in tile
  const int cn = lane & 15;
  const int hi = lane >> 4;

  // Load Q tile (64x128)
  const int lr = tid >> 1;
  const int lc = (tid & 1) * 64;
  {
    const bf16* gq = Q + qkBase + (size_t)(qTile + lr) * HDIM + lc;
#pragma unroll
    for (int i = 0; i < 8; ++i)
      *(v8bf*)&Qs[lr * LQK + lc + i * 8] = *(const v8bf*)(gq + i * 8);
  }

  float m[8], l[8];
  v8f o[8];
#pragma unroll
  for (int r = 0; r < 8; ++r) { m[r] = -3.0e38f; l[r] = 0.0f; }
#pragma unroll
  for (int dt = 0; dt < 8; ++dt) o[dt] = v8f{};

  const int nchunks = qt + 1;
  for (int ch = 0; ch < nchunks; ++ch) {
    const int kb = ch * 64;
    __syncthreads();   // previous K/V tiles no longer in use
    {
      const bf16* gk = Kb + qkBase + (size_t)(kb + lr) * HDIM + lc;
#pragma unroll
      for (int i = 0; i < 8; ++i)
        *(v8bf*)&Ks[lr * LQK + lc + i * 8] = *(const v8bf*)(gk + i * 8);
      const bf16* gv = Vt + qkBase + (size_t)tid * SEQ + kb;  // row d = tid
#pragma unroll
      for (int i = 0; i < 8; ++i)
        *(v8bf*)&Vs[tid * LV + i * 8] = *(const v8bf*)(gv + i * 8);
    }
    __syncthreads();

    // Warm L2/WGP$ for the next chunk while this chunk computes.
    if (ch + 1 < nchunks) {
      __builtin_prefetch(Kb + qkBase + (size_t)(kb + 64 + lr) * HDIM + lc, 0, 0);
      __builtin_prefetch(Vt + qkBase + (size_t)tid * SEQ + kb + 64, 0, 0);
    }

    // Scores: 16 q-rows x 64 keys = 4 n-tiles, K-dim = 128 = 4 steps of 32
    v8f sc[4];
#pragma unroll
    for (int nt = 0; nt < 4; ++nt) sc[nt] = v8f{};
#pragma unroll
    for (int ds = 0; ds < 4; ++ds) {
      v16bf aq = ld_frag_a(&Qs[qW * LQK + ds * 32], LQK, lane);
#pragma unroll
      for (int nt = 0; nt < 4; ++nt) {
        v16bf bk = ld_frag_b(&Ks[(nt * 16) * LQK + ds * 32], LQK, lane);
        sc[nt] = wmma_bf16(aq, bk, sc[nt]);
      }
    }

    // Online softmax (rows live across the 16-lane halves)
    float fac[8];
#pragma unroll
    for (int r = 0; r < 8; ++r) {
      const int qi = qTile + qW + r + hi * 8;
      float mx = m[r];
#pragma unroll
      for (int nt = 0; nt < 4; ++nt) {
        const int kv = kb + nt * 16 + cn;
        float s = sc[nt][r];
        if (kv > qi) s = -3.0e38f;   // causal mask
        sc[nt][r] = s;
        mx = fmaxf(mx, s);
      }
      mx = fmaxf(mx, __shfl_xor(mx, 1, 32));
      mx = fmaxf(mx, __shfl_xor(mx, 2, 32));
      mx = fmaxf(mx, __shfl_xor(mx, 4, 32));
      mx = fmaxf(mx, __shfl_xor(mx, 8, 32));
      fac[r] = exp2f(m[r] - mx);
      m[r] = mx;
      float sum = 0.0f;
#pragma unroll
      for (int nt = 0; nt < 4; ++nt) {
        const float p = exp2f(sc[nt][r] - mx);
        sc[nt][r] = p;
        sum += p;
      }
      sum += __shfl_xor(sum, 1, 32);
      sum += __shfl_xor(sum, 2, 32);
      sum += __shfl_xor(sum, 4, 32);
      sum += __shfl_xor(sum, 8, 32);
      l[r] = l[r] * fac[r] + sum;
    }

    // Rescale running output
#pragma unroll
    for (int dt = 0; dt < 8; ++dt)
#pragma unroll
      for (int r = 0; r < 8; ++r) o[dt][r] *= fac[r];

    // Stage P as bf16 in wave-private LDS (C/D layout -> row-major 16x64)
    bf16* ps = Ps[wave];
#pragma unroll
    for (int nt = 0; nt < 4; ++nt)
#pragma unroll
      for (int r = 0; r < 8; ++r)
        ps[(r + hi * 8) * LV + nt * 16 + cn] = (bf16)sc[nt][r];

    // O += P @ V : K-dim = 64 keys = 2 steps; B operand from transposed V
#pragma unroll
    for (int ks = 0; ks < 2; ++ks) {
      v16bf ap = ld_frag_a(&ps[ks * 32], LV, lane);
#pragma unroll
      for (int dt = 0; dt < 8; ++dt) {
        v16bf bv = ld_frag_b(&Vs[(dt * 16) * LV + ks * 32], LV, lane);
        o[dt] = wmma_bf16(ap, bv, o[dt]);
      }
    }
  }

  // Normalize and scatter to [B,S,E] bf16
  const int h = bh & (HEADS - 1);
  const int b = bh >> 4;
#pragma unroll
  for (int r = 0; r < 8; ++r) {
    const float rl = 1.0f / l[r];
    const int q = qTile + qW + r + hi * 8;
    const size_t rowBase = ((size_t)(b * SEQ + q)) * EMBED + h * HDIM;
#pragma unroll
    for (int dt = 0; dt < 8; ++dt)
      O[rowBase + dt * 16 + cn] = (bf16)(o[dt][r] * rl);
  }
}

// ---------------------------------------------------------------------------
// Launcher
// ---------------------------------------------------------------------------
extern "C" void kernel_launch(void* const* d_in, const int* in_sizes, int n_in,
                              void* d_out, int out_size, void* d_ws, size_t ws_size,
                              hipStream_t stream) {
  const float* x  = (const float*)d_in[0];
  const float* Wq = (const float*)d_in[1];
  const float* bq = (const float*)d_in[2];
  const float* Wk = (const float*)d_in[3];
  const float* bk = (const float*)d_in[4];
  const float* Wv = (const float*)d_in[5];
  const float* bv = (const float*)d_in[6];
  const float* Wo = (const float*)d_in[7];
  const float* bo = (const float*)d_in[8];
  float* out = (float*)d_out;

  char* ws = (char*)d_ws;
  size_t off = 0;
  const size_t szX = (size_t)MROWS * EMBED * sizeof(bf16);
  const size_t szW = (size_t)EMBED * EMBED * sizeof(bf16);
  const size_t szH = (size_t)BATCH * HEADS * SEQ * HDIM * sizeof(bf16);
  bf16* xb  = (bf16*)(ws + off); off += szX;
  bf16* wqb = (bf16*)(ws + off); off += szW;
  bf16* wkb = (bf16*)(ws + off); off += szW;
  bf16* wvb = (bf16*)(ws + off); off += szW;
  bf16* wob = (bf16*)(ws + off); off += szW;
  bf16* qb  = (bf16*)(ws + off); off += szH;
  bf16* kb2 = (bf16*)(ws + off); off += szH;
  bf16* vtb = (bf16*)(ws + off); off += szH;
  bf16* aob = (bf16*)(ws + off); off += szX;

  // f32 -> bf16 conversions
  {
    int n4 = MROWS * EMBED / 4;
    cvt_f32_bf16<<<(n4 + 255) / 256, 256, 0, stream>>>(x, xb, n4);
    n4 = EMBED * EMBED / 4;
    cvt_f32_bf16<<<(n4 + 255) / 256, 256, 0, stream>>>(Wq, wqb, n4);
    cvt_f32_bf16<<<(n4 + 255) / 256, 256, 0, stream>>>(Wk, wkb, n4);
    cvt_f32_bf16<<<(n4 + 255) / 256, 256, 0, stream>>>(Wv, wvb, n4);
    cvt_f32_bf16<<<(n4 + 255) / 256, 256, 0, stream>>>(Wo, wob, n4);
  }

  // Q/K/V projections (WMMA GEMM). Q pre-scaled by log2(e)/sqrt(HDIM) so the
  // softmax can run on exp2.
  const float qscale = 1.4426950408889634f * 0.08838834764831845f;
  dim3 gg(EMBED / 128, MROWS / 128);
  gemm_wmma_bf16<<<gg, 256, 0, stream>>>(xb, wqb, bq, nullptr, qb,
                                         MROWS, EMBED, EMBED, 1, qscale);
  gemm_wmma_bf16<<<gg, 256, 0, stream>>>(xb, wkb, bk, nullptr, kb2,
                                         MROWS, EMBED, EMBED, 1, 1.0f);
  gemm_wmma_bf16<<<gg, 256, 0, stream>>>(xb, wvb, bv, nullptr, vtb,
                                         MROWS, EMBED, EMBED, 2, 1.0f);

  // Causal flash attention
  flash_wmma_bf16<<<BATCH * HEADS * (SEQ / 64), 128, 0, stream>>>(qb, kb2, vtb, aob);

  // Output projection -> f32 d_out
  gemm_wmma_bf16<<<gg, 256, 0, stream>>>(aob, wob, bo, out, nullptr,
                                         MROWS, EMBED, EMBED, 0, 1.0f);
}
